// MultiHeadLatentAttention_10479720203080
// MI455X (gfx1250) — compile-verified, matrix operand order checked by reading, not run on hardware
//
#include <hip/hip_runtime.h>
#include <hip/hip_bf16.h>

// ---------------------------------------------------------------------------
// MLA forward for MI455X (gfx1250, wave32). All matmuls via
// v_wmma_f32_16x16x32_bf16 (bf16 A/B, fp32 accumulate). Flash-style
// attention (online softmax) so the 256MB T x T att matrix is never
// materialized. GEMM tile staging uses the CDNA5 Tensor Data Mover
// (tensor_load_to_lds, TENSORcnt) with double-buffered LDS when the
// builtin is available; manual global->LDS staging otherwise.
// Shapes are compile-time: B=4, T=2048, C=1024, H=16, L=32.
// ---------------------------------------------------------------------------

typedef __attribute__((ext_vector_type(16))) __bf16 v16bf;
typedef __attribute__((ext_vector_type(8)))  float  v8f;
typedef __attribute__((ext_vector_type(4)))  unsigned int v4u;
typedef __attribute__((ext_vector_type(8)))  int v8i_;
typedef __attribute__((ext_vector_type(4)))  int v4i_;

#define B_SZ   4
#define T_SZ   2048
#define C_SZ   1024
#define H_SZ   16
#define HD_SZ  64
#define L_SZ   32
#define NEGBIG (-3.0e38f)

#if __has_builtin(__builtin_amdgcn_tensor_load_to_lds) && \
    __has_builtin(__builtin_amdgcn_s_wait_tensorcnt)
#define HAVE_TDM 1
#else
#define HAVE_TDM 0
#endif
#if __has_include(<hip/amd_detail/amd_gfx1250_TDM.h>)
#define TDM_6ARG 1   // amdgpu-toolchain (clang-23): 6-arg builtin
#else
#define TDM_6ARG 0   // ROCm 7.2 (clang-22): 5-arg builtin
#endif

union Frag16 { v16bf v; uint4 q[2]; __bf16 e[16]; };

// CDNA5 16-bit A/B fragment loader for row-major storage:
//   lane<16 : row r=lane,    k in {0..7} then {16..23}
//   lane>=16: row r=lane-16, k in {8..15} then {24..31}
// => per lane: 16B at (r*ld + half*8), 16B at +16 elements.
__device__ __forceinline__ v16bf load_frag(const __bf16* base, int row0, int ld) {
  int lane = threadIdx.x & 31;
  int half = lane >> 4, r = lane & 15;
  const __bf16* p = base + (size_t)(row0 + r) * ld + half * 8;
  Frag16 f;
  f.q[0] = *(const uint4*)(p);
  f.q[1] = *(const uint4*)(p + 16);
  return f.v;
}

__device__ __forceinline__ v8f wmma_bf16(v16bf a, v16bf b, v8f c) {
  return __builtin_amdgcn_wmma_f32_16x16x32_bf16(false, a, false, b, (short)0, c,
                                                 false, false);
}

#if HAVE_TDM
// TDM: async-load one bf16 tile of `rows` x 32 elems (row stride K elems) from
// global into LDS at byte offset lds_off, inserting 4 DWORDs of pad after each
// 16 DWORDs (=> 40-element LDS row stride matching load_frag). Descriptor per
// CDNA5 ISA ch.8: exact-fit tensor dims (no OOB), 2-D (groups 2/3 zero).
__device__ __forceinline__ void tdm_tile_load(const __bf16* gptr, unsigned lds_off,
                                              int rows, long k_stride) {
  unsigned long long ga = (unsigned long long)(uintptr_t)gptr;
  v4u g0;
  g0.x = 1u;                                   // count=1, no gather
  g0.y = lds_off;                              // lds_addr
  g0.z = (unsigned)ga;                         // global_addr[31:0]
  g0.w = (unsigned)((ga >> 32) & 0x01FFFFFFu)  // global_addr[56:32]
         | (2u << 30);                         // type=2 ("image")
  v8i_ g1;
  g1[0] = (int)((1u << 16)      // data_size = 2 bytes
              | (1u << 20)      // pad_enable
              | (3u << 22)      // pad_interval: every 16 DWORDs
              | (3u << 25));    // pad_amount: 4 DWORDs
  unsigned td0 = 32u, td1 = (unsigned)rows;
  g1[1] = (int)((td0 & 0xFFFFu) << 16);                    // [63:48] dim0 lo
  g1[2] = (int)((td0 >> 16) | ((td1 & 0xFFFFu) << 16));    // dim0 hi | dim1 lo
  g1[3] = (int)((td1 >> 16) | (32u << 16));                // dim1 hi | tile_dim0
  g1[4] = (int)((unsigned)rows & 0xFFFFu);                 // tile_dim1 (dim2=0)
  g1[5] = (int)(unsigned)(k_stride & 0xFFFFFFFF);          // dim0_stride lo
  g1[6] = (int)(unsigned)((k_stride >> 32) & 0xFFFF);      // dim0_stride hi
  g1[7] = 0;
  v4i_ z4 = {0, 0, 0, 0};
#if TDM_6ARG
  v8i_ z8 = {0, 0, 0, 0, 0, 0, 0, 0};
  __builtin_amdgcn_tensor_load_to_lds(g0, g1, z4, z4, z8, 0);
#else
  __builtin_amdgcn_tensor_load_to_lds(g0, g1, z4, z4, 0);
#endif
}
#endif

// ---------------------------------------------------------------------------
// f32 -> bf16 conversion
// ---------------------------------------------------------------------------
__global__ __launch_bounds__(256) void cvt_f32_to_bf16(const float* __restrict__ in,
                                                       __bf16* __restrict__ out, int n) {
  int i = blockIdx.x * 256 + threadIdx.x;
  if (i < n) out[i] = (__bf16)in[i];
}

// ---------------------------------------------------------------------------
// Tiled GEMM: C[M,N] = A[M,K](bf16) @ W[N,K]^T(bf16) + bias, fp32 out.
// Block = 256 threads (8 waves). Tile 128x128, K-step 32, double-buffered LDS
// filled by the Tensor Data Mover (waves 0/1 issue, TENSORcnt sync).
// Wave (wm,wn) in 2x4 grid computes a 64x32 sub-tile = 4x2 WMMA tiles.
// ---------------------------------------------------------------------------
__global__ __launch_bounds__(256)
void gemm_bf16_wmma(const __bf16* __restrict__ A, const __bf16* __restrict__ W,
                    const float* __restrict__ bias, float* __restrict__ C,
                    int M, int N, int K) {
  __shared__ __bf16 As[2][128 * 40];   // 40 = 32 + 8 pad (TDM pad / 16B align)
  __shared__ __bf16 Bs[2][128 * 40];

  const int tid  = threadIdx.x;
  const int lane = tid & 31;
  const int w    = tid >> 5;
  const int wm   = w >> 2;   // 0..1
  const int wn   = w & 3;    // 0..3
  const int m0   = blockIdx.y * 128;
  const int n0   = blockIdx.x * 128;
  const int nk   = K >> 5;

  v8f acc[4][2];
#pragma unroll
  for (int i = 0; i < 4; ++i)
#pragma unroll
    for (int j = 0; j < 2; ++j) acc[i][j] = (v8f){0.f, 0.f, 0.f, 0.f, 0.f, 0.f, 0.f, 0.f};

#if HAVE_TDM
  // Wave 0 streams A tiles, wave 1 streams B tiles, all async via TDM.
  if (w < 2) {
    const __bf16* src = (w == 0) ? A + (size_t)m0 * K : W + (size_t)n0 * K;
    unsigned dst = (unsigned)(uintptr_t)((w == 0) ? &As[0][0] : &Bs[0][0]);
    tdm_tile_load(src, dst, 128, K);
  }
  for (int ki = 0; ki < nk; ++ki) {
    const int buf = ki & 1;
    if (w < 2) {
      if (ki + 1 < nk) {
        const __bf16* src = (w == 0) ? A + (size_t)m0 * K + (ki + 1) * 32
                                     : W + (size_t)n0 * K + (ki + 1) * 32;
        unsigned dst =
            (unsigned)(uintptr_t)((w == 0) ? &As[buf ^ 1][0] : &Bs[buf ^ 1][0]);
        tdm_tile_load(src, dst, 128, K);
        __builtin_amdgcn_s_wait_tensorcnt(1);  // oldest tile (ki) landed
      } else {
        __builtin_amdgcn_s_wait_tensorcnt(0);
      }
    }
    __syncthreads();

    v16bf af[4], bf[2];
#pragma unroll
    for (int mt = 0; mt < 4; ++mt) af[mt] = load_frag(&As[buf][0], wm * 64 + mt * 16, 40);
#pragma unroll
    for (int nt = 0; nt < 2; ++nt) bf[nt] = load_frag(&Bs[buf][0], wn * 32 + nt * 16, 40);
#pragma unroll
    for (int mt = 0; mt < 4; ++mt)
#pragma unroll
      for (int nt = 0; nt < 2; ++nt) acc[mt][nt] = wmma_bf16(af[mt], bf[nt], acc[mt][nt]);
    __syncthreads();   // readers done before TDM refills this buffer
  }
#else
  for (int ki = 0; ki < nk; ++ki) {
    const int k0 = ki * 32;
#pragma unroll
    for (int i = 0; i < 2; ++i) {
      int idx = tid + i * 256;
      int row = idx >> 2;
      int ch  = (idx & 3) * 8;
      *(uint4*)&As[0][row * 40 + ch] = *(const uint4*)&A[(size_t)(m0 + row) * K + k0 + ch];
      *(uint4*)&Bs[0][row * 40 + ch] = *(const uint4*)&W[(size_t)(n0 + row) * K + k0 + ch];
    }
    if (k0 + 32 < K) {
      __builtin_prefetch(&A[(size_t)(m0 + (tid >> 1)) * K + k0 + 32], 0, 3);
      __builtin_prefetch(&W[(size_t)(n0 + (tid >> 1)) * K + k0 + 32], 0, 3);
    }
    __syncthreads();
    v16bf af[4], bf[2];
#pragma unroll
    for (int mt = 0; mt < 4; ++mt) af[mt] = load_frag(&As[0][0], wm * 64 + mt * 16, 40);
#pragma unroll
    for (int nt = 0; nt < 2; ++nt) bf[nt] = load_frag(&Bs[0][0], wn * 32 + nt * 16, 40);
#pragma unroll
    for (int mt = 0; mt < 4; ++mt)
#pragma unroll
      for (int nt = 0; nt < 2; ++nt) acc[mt][nt] = wmma_bf16(af[mt], bf[nt], acc[mt][nt]);
    __syncthreads();
  }
#endif

  // Epilogue: C/D layout — lane holds col = lane&15, rows j + 8*(lane>>4).
  const int half = lane >> 4, col = lane & 15;
#pragma unroll
  for (int mt = 0; mt < 4; ++mt)
#pragma unroll
    for (int nt = 0; nt < 2; ++nt) {
      int cc = n0 + wn * 32 + nt * 16 + col;
      float bv = bias[cc];
#pragma unroll
      for (int j = 0; j < 8; ++j) {
        int rr = m0 + wm * 64 + mt * 16 + j + 8 * half;
        C[(size_t)rr * N + cc] = acc[mt][nt][j] + bv;
      }
    }
}

// ---------------------------------------------------------------------------
// Per-token latent projections + RoPE + wqk fold (tiny FLOPs, VALU f32).
// Emits: qf = (wqk @ rope(q2l@q)) [B,H,T,L] bf16 (A-frag friendly)
//        kl = rope(kv2l@k)        [B,H,T,L] bf16 (B-frag friendly for S)
//        vt = (kv2l@v)^T          [B,H,L,T] bf16 (B-frag friendly for PV)
// ---------------------------------------------------------------------------
__device__ __forceinline__ void rope32(float* lat, int t) {
#pragma unroll
  for (int j = 0; j < 16; ++j) {
    float invf = __powf(10000.f, -(2.f * j) / 32.f);
    float s, c;
    __sincosf((float)t * invf, &s, &c);
    float e = lat[2 * j], o = lat[2 * j + 1];
    lat[2 * j]     = e * c - o * s;
    lat[2 * j + 1] = o * c + e * s;
  }
}

__global__ __launch_bounds__(256)
void latent_rope_kernel(const float* __restrict__ qkv,
                        const float* __restrict__ q2l_w, const float* __restrict__ q2l_b,
                        const float* __restrict__ kv2l_w, const float* __restrict__ kv2l_b,
                        const float* __restrict__ wqk_w, const float* __restrict__ wqk_b,
                        __bf16* __restrict__ qf, __bf16* __restrict__ kl,
                        __bf16* __restrict__ vt) {
  __shared__ float wq[32 * 64], wkv[32 * 64], wk[32 * 32], bq[32], bkv[32], bk[32];
  for (int i = threadIdx.x; i < 32 * 64; i += 256) { wq[i] = q2l_w[i]; wkv[i] = kv2l_w[i]; }
  for (int i = threadIdx.x; i < 32 * 32; i += 256) wk[i] = wqk_w[i];
  if (threadIdx.x < 32) {
    bq[threadIdx.x]  = q2l_b[threadIdx.x];
    bkv[threadIdx.x] = kv2l_b[threadIdx.x];
    bk[threadIdx.x]  = wqk_b[threadIdx.x];
  }
  __syncthreads();

  int idx = blockIdx.x * 256 + threadIdx.x;        // over B*H*T = 2^17
  int t = idx & (T_SZ - 1);
  int h = (idx >> 11) & (H_SZ - 1);
  int b = idx >> 15;
  size_t base = ((size_t)(b * T_SZ + t)) * (3 * C_SZ) + h * HD_SZ;
  size_t ob   = ((size_t)((b * H_SZ + h) * T_SZ + t)) * L_SZ;

  float xin[64], lat[32];

  // ---- Q: q2l -> rope -> wqk ----
#pragma unroll
  for (int i = 0; i < 64; ++i) xin[i] = qkv[base + i];
#pragma unroll
  for (int l = 0; l < 32; ++l) {
    float s = bq[l];
#pragma unroll
    for (int i = 0; i < 64; ++i) s += xin[i] * wq[l * 64 + i];
    lat[l] = s;
  }
  rope32(lat, t);
#pragma unroll
  for (int l = 0; l < 32; ++l) {
    float s = bk[l];
#pragma unroll
    for (int m = 0; m < 32; ++m) s += wk[l * 32 + m] * lat[m];
    qf[ob + l] = (__bf16)s;
  }

  // ---- K: kv2l -> rope ----
#pragma unroll
  for (int i = 0; i < 64; ++i) xin[i] = qkv[base + C_SZ + i];
#pragma unroll
  for (int l = 0; l < 32; ++l) {
    float s = bkv[l];
#pragma unroll
    for (int i = 0; i < 64; ++i) s += xin[i] * wkv[l * 64 + i];
    lat[l] = s;
  }
  rope32(lat, t);
#pragma unroll
  for (int l = 0; l < 32; ++l) kl[ob + l] = (__bf16)lat[l];

  // ---- V: kv2l (no rope), stored transposed [L,T] ----
#pragma unroll
  for (int i = 0; i < 64; ++i) xin[i] = qkv[base + 2 * C_SZ + i];
  size_t vb = ((size_t)(b * H_SZ + h)) * L_SZ * T_SZ;
#pragma unroll
  for (int l = 0; l < 32; ++l) {
    float s = bkv[l];
#pragma unroll
    for (int i = 0; i < 64; ++i) s += xin[i] * wkv[l * 64 + i];
    vt[vb + (size_t)l * T_SZ + t] = (__bf16)s;
  }
}

// ---------------------------------------------------------------------------
// Flash attention + fused l2o. One wave per 16-query tile, 4 waves/block.
// Per 32-key block: 2 WMMAs for S (K=L=32), online softmax with 16-lane
// shfl reductions, P transposed C->A layout through LDS, 2 WMMAs for PV.
// Epilogue: y_lat -> l2o (4 WMMAs) -> y bf16 [B,T,C].
// ---------------------------------------------------------------------------
__global__ __launch_bounds__(128)
void mla_flash_kernel(const __bf16* __restrict__ qf, const __bf16* __restrict__ kl,
                      const __bf16* __restrict__ vt, const __bf16* __restrict__ l2o,
                      const float* __restrict__ l2o_b, __bf16* __restrict__ ybf) {
  __shared__ __bf16 stage[4][16 * 40];   // per-wave 16x32 (+pad) bf16 scratch
  const int wid  = threadIdx.x >> 5;
  const int lane = threadIdx.x & 31;
  const int half = lane >> 4, col = lane & 15;

  int tile = blockIdx.x * 4 + wid;        // over B*H*(T/16) = 8192
  int qt = tile & 127;
  int bh = tile >> 7;
  int b = bh >> 4, h = bh & 15;
  int qbase = qt * 16;

  const __bf16* qbp = qf + ((size_t)bh * T_SZ + qbase) * L_SZ;
  const __bf16* kbp = kl + (size_t)bh * T_SZ * L_SZ;
  const __bf16* vbp = vt + (size_t)bh * L_SZ * T_SZ;
  __bf16* st = &stage[wid][0];

  const v16bf qA = load_frag(qbp, 0, L_SZ);
  const v8f zero8 = (v8f){0.f, 0.f, 0.f, 0.f, 0.f, 0.f, 0.f, 0.f};
  v8f acc0 = zero8, acc1 = zero8;   // y_lat latent [0..15] / [16..31]
  float mrun[8], lrun[8];
#pragma unroll
  for (int j = 0; j < 8; ++j) { mrun[j] = NEGBIG; lrun[j] = 0.f; }

  const float sc = 0.17677669529663687f;  // L^-0.5
  const int nkb = (qbase + 16 + 31) >> 5;

  for (int kbI = 0; kbI < nkb; ++kbI) {
    const int kb = kbI * 32;
    v16bf kB0 = load_frag(kbp + (size_t)kb * L_SZ, 0, L_SZ);
    v16bf kB1 = load_frag(kbp + (size_t)(kb + 16) * L_SZ, 0, L_SZ);
    v8f S0 = wmma_bf16(qA, kB0, zero8);
    v8f S1 = wmma_bf16(qA, kB1, zero8);

#pragma unroll
    for (int j = 0; j < 8; ++j) {
      const int m = j + 8 * half;           // local query row
      const int qidx = qbase + m;
      float s0 = S0[j] * sc, s1 = S1[j] * sc;
      if (kb + col > qidx)      s0 = NEGBIG;   // causal mask
      if (kb + 16 + col > qidx) s1 = NEGBIG;
      float rm = fmaxf(s0, s1);
      rm = fmaxf(rm, __shfl_xor(rm, 1));
      rm = fmaxf(rm, __shfl_xor(rm, 2));
      rm = fmaxf(rm, __shfl_xor(rm, 4));
      rm = fmaxf(rm, __shfl_xor(rm, 8));
      float mnew = fmaxf(mrun[j], rm);
      float p0 = __expf(s0 - mnew), p1 = __expf(s1 - mnew);
      float rs = p0 + p1;
      rs += __shfl_xor(rs, 1);
      rs += __shfl_xor(rs, 2);
      rs += __shfl_xor(rs, 4);
      rs += __shfl_xor(rs, 8);
      float resc = __expf(mrun[j] - mnew);
      lrun[j] = lrun[j] * resc + rs;
      mrun[j] = mnew;
      acc0[j] *= resc;
      acc1[j] *= resc;
      st[m * 40 + col]      = (__bf16)p0;   // stage P (C-layout -> row-major)
      st[m * 40 + 16 + col] = (__bf16)p1;
    }
    asm volatile("s_wait_dscnt 0" ::: "memory");   // wave-local DS RAW
    v16bf pA  = load_frag(st, 0, 40);              // P as A-fragment (K=32 keys)
    v16bf vB0 = load_frag(vbp + kb, 0, T_SZ);               // latent 0..15
    v16bf vB1 = load_frag(vbp + 16 * T_SZ + kb, 0, T_SZ);   // latent 16..31
    acc0 = wmma_bf16(pA, vB0, acc0);
    acc1 = wmma_bf16(pA, vB1, acc1);
  }

  // Normalize and stage y_lat (16x32) for the l2o GEMM.
#pragma unroll
  for (int j = 0; j < 8; ++j) {
    const int m = j + 8 * half;
    float inv = lrun[j] > 0.f ? 1.f / lrun[j] : 0.f;
    st[m * 40 + col]      = (__bf16)(acc0[j] * inv);
    st[m * 40 + 16 + col] = (__bf16)(acc1[j] * inv);
  }
  asm volatile("s_wait_dscnt 0" ::: "memory");
  v16bf yA = load_frag(st, 0, 40);
#pragma unroll
  for (int ot = 0; ot < 4; ++ot) {                  // 64 output channels / 16
    v16bf oB = load_frag(l2o, ot * 16, L_SZ);
    v8f o = wmma_bf16(yA, oB, zero8);
    float bv = l2o_b[ot * 16 + col];
#pragma unroll
    for (int j = 0; j < 8; ++j) {
      int trow = qbase + j + 8 * half;
      ybf[((size_t)(b * T_SZ + trow)) * C_SZ + h * HD_SZ + ot * 16 + col] =
          (__bf16)(o[j] + bv);
    }
  }
}

// ---------------------------------------------------------------------------
// Launch
// ---------------------------------------------------------------------------
extern "C" void kernel_launch(void* const* d_in, const int* in_sizes, int n_in,
                              void* d_out, int out_size, void* d_ws, size_t ws_size,
                              hipStream_t stream) {
  const float* x        = (const float*)d_in[0];
  const float* c_attn_w = (const float*)d_in[1];
  const float* c_attn_b = (const float*)d_in[2];
  const float* q2l_w    = (const float*)d_in[3];
  const float* q2l_b    = (const float*)d_in[4];
  const float* kv2l_w   = (const float*)d_in[5];
  const float* kv2l_b   = (const float*)d_in[6];
  const float* l2o_w    = (const float*)d_in[7];
  const float* l2o_b    = (const float*)d_in[8];
  const float* wqk_w    = (const float*)d_in[9];
  const float* wqk_b    = (const float*)d_in[10];
  const float* cproj_w  = (const float*)d_in[11];
  const float* cproj_b  = (const float*)d_in[12];
  float* out = (float*)d_out;

  char* ws = (char*)d_ws;
  size_t off = 0;
  auto carve = [&](size_t bytes) -> void* {
    void* p = ws + off;
    off += (bytes + 255) & ~(size_t)255;
    return p;
  };
  const size_t BT = (size_t)B_SZ * T_SZ;                 // 8192 rows
  float*  qkv   = (float*)carve(BT * 3 * C_SZ * 4);      //  96 MB fp32
  __bf16* x_bf  = (__bf16*)carve(BT * C_SZ * 2);         //  16 MB
  __bf16* wa_bf = (__bf16*)carve((size_t)3 * C_SZ * C_SZ * 2);  // 6 MB
  __bf16* wp_bf = (__bf16*)carve((size_t)C_SZ * C_SZ * 2);      // 2 MB
  __bf16* lo_bf = (__bf16*)carve((size_t)HD_SZ * L_SZ * 2);
  __bf16* qf    = (__bf16*)carve(BT * H_SZ * L_SZ * 2);  // 8 MB [B,H,T,L]
  __bf16* klat  = (__bf16*)carve(BT * H_SZ * L_SZ * 2);  // 8 MB [B,H,T,L]
  __bf16* vlt   = (__bf16*)carve(BT * H_SZ * L_SZ * 2);  // 8 MB [B,H,L,T]
  __bf16* y_bf  = (__bf16*)carve(BT * C_SZ * 2);         // 16 MB

  // 1) bf16 conversions
  int nx = (int)(BT * C_SZ);
  cvt_f32_to_bf16<<<(nx + 255) / 256, 256, 0, stream>>>(x, x_bf, nx);
  int nwa = 3 * C_SZ * C_SZ;
  cvt_f32_to_bf16<<<(nwa + 255) / 256, 256, 0, stream>>>(c_attn_w, wa_bf, nwa);
  int nwp = C_SZ * C_SZ;
  cvt_f32_to_bf16<<<(nwp + 255) / 256, 256, 0, stream>>>(cproj_w, wp_bf, nwp);
  cvt_f32_to_bf16<<<(HD_SZ * L_SZ + 255) / 256, 256, 0, stream>>>(l2o_w, lo_bf,
                                                                  HD_SZ * L_SZ);

  // 2) qkv = x @ c_attn_w^T + b   (8192 x 3072 x 1024)
  gemm_bf16_wmma<<<dim3(3 * C_SZ / 128, (int)(BT / 128)), 256, 0, stream>>>(
      x_bf, wa_bf, c_attn_b, qkv, (int)BT, 3 * C_SZ, C_SZ);

  // 3) latent projections + RoPE + wqk
  latent_rope_kernel<<<(B_SZ * H_SZ * T_SZ) / 256, 256, 0, stream>>>(
      qkv, q2l_w, q2l_b, kv2l_w, kv2l_b, wqk_w, wqk_b, qf, klat, vlt);

  // 4) flash attention + l2o (8192 query tiles, 4 waves/block)
  mla_flash_kernel<<<(B_SZ * H_SZ * (T_SZ / 16)) / 4, 128, 0, stream>>>(
      qf, klat, vlt, lo_bf, l2o_b, y_bf);

  // 5) out = y @ cproj_w^T + b   (8192 x 1024 x 1024)
  gemm_bf16_wmma<<<dim3(C_SZ / 128, (int)(BT / 128)), 256, 0, stream>>>(
      y_bf, wp_bf, cproj_b, out, (int)BT, C_SZ, C_SZ);

  (void)in_sizes; (void)n_in; (void)out_size; (void)ws_size;
}
